// BEMv11Module_57226144252173
// MI455X (gfx1250) — matrix operand order repaired and checked by reading, not compiled, not run on hardware
//
#include <hip/hip_runtime.h>

typedef __attribute__((ext_vector_type(16))) __bf16 v16bf;
typedef __attribute__((ext_vector_type(2)))  __bf16 bf2;
typedef __attribute__((ext_vector_type(8)))  float  v8f;
typedef __attribute__((ext_vector_type(4)))  float  f4;
typedef __attribute__((ext_vector_type(2)))  float  f2;
typedef unsigned int u32;
typedef __attribute__((ext_vector_type(4))) u32 u32x4;
typedef __attribute__((ext_vector_type(2))) u32 u32x2;

#define D_IN    4096
#define D_OUT   4096
#define D_HID   2048
#define SEQ     4096
#define BATCH   2
#define NTOK    (BATCH * SEQ)      // 8192
#define RANK    16
#define NEXP    8
#define CHUNK   128
#define NCHUNK  (NTOK / CHUNK)     // 64
#define CPB     (SEQ / CHUNK)      // 32 chunks per batch
#define TAU     0.7f
#define SCALING 1.0f               // ALPHA / RANK = 16/16
#define XSW     20                 // dwords per LDS tile row (16 used + 4 pad)

union frag_u { v16bf v; u32x4 q[2]; };
union pk_u   { bf2 b; u32 u; };

__device__ __forceinline__ u32 pack_hi2(float a, float b) {
    f2 v = {a, b};
    pk_u t;
    t.b = __builtin_convertvector(v, bf2);        // packed v_cvt_pk_bf16_f32
    return t.u;
}
__device__ __forceinline__ u32 pack_lo2(float a, float b) {
    f2 v = {a, b};
    bf2 h  = __builtin_convertvector(v, bf2);
    f2  hf = __builtin_convertvector(h, f2);
    f2  r  = v - hf;
    pk_u t;
    t.b = __builtin_convertvector(r, bf2);
    return t.u;
}

// ---------------------------------------------------------------------------
// Kernel 1: per-chunk column sums of relu(x @ W1 + b1).
// Grid: (NCHUNK, D_HID/128). Block: 256 = 8 waves (2 M x 4 N).
// Single-pass bf16 (router only feeds argmax + 0.7 hysteresis threshold).
// LDS-staged tiles: x[128x32] row-major, W1[32x128] stored transposed [col][k].
// ---------------------------------------------------------------------------
__global__ __launch_bounds__(256) void router_hidden_kernel(
    const float* __restrict__ x, const float* __restrict__ W1,
    const float* __restrict__ b1, float* __restrict__ chunkSum)
{
    const int chunk = blockIdx.x;
    const int htile = blockIdx.y;
    const int tid   = threadIdx.x;
    const int lane  = tid & 31;
    const int wave  = tid >> 5;
    const int wm    = wave >> 2;
    const int wn    = wave & 3;
    const int lhalf = lane >> 4;
    const int lmod  = lane & 15;

    __shared__ u32 xs_hi[2][128][XSW];
    __shared__ u32 wt_hi[2][128][XSW];
    __shared__ float colsum[128];

    auto stage = [&](int nb, int kk) {
        {   // x tile: thread -> (row, 16-wide k half)
            const int row = tid >> 1;
            const int kh  = (tid & 1) * 16;
            const float* p = x + (size_t)(chunk * CHUNK + row) * D_IN + kk + kh;
            f4 f0 = *(const f4*)(p);
            f4 f1 = *(const f4*)(p + 4);
            f4 f2v = *(const f4*)(p + 8);
            f4 f3 = *(const f4*)(p + 12);
            u32x4 h0, h1;
            h0[0] = pack_hi2(f0[0], f0[1]); h0[1] = pack_hi2(f0[2], f0[3]);
            h0[2] = pack_hi2(f1[0], f1[1]); h0[3] = pack_hi2(f1[2], f1[3]);
            h1[0] = pack_hi2(f2v[0], f2v[1]); h1[1] = pack_hi2(f2v[2], f2v[3]);
            h1[2] = pack_hi2(f3[0], f3[1]); h1[3] = pack_hi2(f3[2], f3[3]);
            *(u32x4*)&xs_hi[nb][row][kh >> 1]       = h0;
            *(u32x4*)&xs_hi[nb][row][(kh >> 1) + 4] = h1;
        }
        {   // W1 tile: thread -> 4x4 micro-block, stored transposed [col][k]
            const int kb = tid & 7;     // 4 k-rows each
            const int cb = tid >> 3;    // 4 cols each
            const float* wp = W1 + (size_t)(kk + kb * 4) * D_HID + htile * 128 + cb * 4;
            f4 w0 = *(const f4*)(wp);
            f4 w1 = *(const f4*)(wp + D_HID);
            f4 w2 = *(const f4*)(wp + 2 * D_HID);
            f4 w3 = *(const f4*)(wp + 3 * D_HID);
            #pragma unroll
            for (int j = 0; j < 4; ++j) {
                u32x2 h;
                h[0] = pack_hi2(w0[j], w1[j]);
                h[1] = pack_hi2(w2[j], w3[j]);
                *(u32x2*)&wt_hi[nb][cb * 4 + j][kb * 2] = h;
            }
        }
    };

    v8f acc[4][2];
    #pragma unroll
    for (int a = 0; a < 4; ++a)
        #pragma unroll
        for (int b = 0; b < 2; ++b)
            #pragma unroll
            for (int i = 0; i < 8; ++i) acc[a][b][i] = 0.f;

    if (tid < 128) colsum[tid] = 0.f;
    stage(0, 0);

    for (int k0 = 0; k0 < D_IN; k0 += 32) {
        const int cbuf = (k0 >> 5) & 1;
        __syncthreads();
        if (k0 + 32 < D_IN) stage(cbuf ^ 1, k0 + 32);

        frag_u B[2];
        #pragma unroll
        for (int tn = 0; tn < 2; ++tn) {
            const int col = wn * 32 + tn * 16 + lmod;
            B[tn].q[0] = *(const u32x4*)&wt_hi[cbuf][col][lhalf * 8];
            B[tn].q[1] = *(const u32x4*)&wt_hi[cbuf][col][lhalf * 8 + 4];
        }
        #pragma unroll
        for (int tm = 0; tm < 4; ++tm) {
            const int row = wm * 64 + tm * 16 + lmod;
            frag_u A;
            A.q[0] = *(const u32x4*)&xs_hi[cbuf][row][lhalf * 8];
            A.q[1] = *(const u32x4*)&xs_hi[cbuf][row][lhalf * 8 + 4];
            #pragma unroll
            for (int tn = 0; tn < 2; ++tn)
                acc[tm][tn] = __builtin_amdgcn_wmma_f32_16x16x32_bf16(
                    false, A.v, false, B[tn].v, (short)0, acc[tm][tn], false, false);
        }
    }
    __syncthreads();

    #pragma unroll
    for (int tn = 0; tn < 2; ++tn) {
        const int col = htile * 128 + wn * 32 + tn * 16 + lmod;
        const float bias = b1[col];
        float s = 0.f;
        #pragma unroll
        for (int tm = 0; tm < 4; ++tm)
            #pragma unroll
            for (int i = 0; i < 8; ++i) {
                float v = acc[tm][tn][i] + bias;
                s += v > 0.f ? v : 0.f;
            }
        s += __shfl_xor(s, 16, 32);          // combine the two M-halves per column
        if (lhalf == 0) atomicAdd(&colsum[wn * 32 + tn * 16 + lmod], s);
    }
    __syncthreads();
    if (tid < 128)
        chunkSum[(size_t)chunk * D_HID + htile * 128 + tid] = colsum[tid];
}

// ---------------------------------------------------------------------------
// Kernel 2: chunk logits = (chunkSum/128) @ W2 + b2, then sequential
// hysteresis routing scan. Single block.
// ---------------------------------------------------------------------------
__global__ __launch_bounds__(256) void router_logits_scan_kernel(
    const float* __restrict__ chunkSum, const float* __restrict__ W2,
    const float* __restrict__ b2, int* __restrict__ expert_idx)
{
    __shared__ float cl[NCHUNK][NEXP];
    const int tid = threadIdx.x;
    for (int p = tid; p < NCHUNK * NEXP; p += 256) {
        const int c = p / NEXP, e = p % NEXP;
        const float* cs = chunkSum + (size_t)c * D_HID;
        float acc = 0.f;
        for (int k = 0; k < D_HID; ++k) acc += cs[k] * W2[k * NEXP + e];
        cl[c][e] = acc * (1.f / CHUNK) + b2[e];
    }
    __syncthreads();
    if (tid < BATCH) {
        const int base = tid * CPB;
        int prev = 0;
        float mv = cl[base][0];
        for (int e = 1; e < NEXP; ++e)
            if (cl[base][e] > mv) { mv = cl[base][e]; prev = e; }
        expert_idx[base] = prev;
        for (int c = 1; c < CPB; ++c) {
            int cand = 0;
            float cv = cl[base + c][0];
            for (int e = 1; e < NEXP; ++e)
                if (cl[base + c][e] > cv) { cv = cl[base + c][e]; cand = e; }
            if (cv - cl[base + c][prev] > TAU) prev = cand;
            expert_idx[base + c] = prev;
        }
    }
}

// ---------------------------------------------------------------------------
// Kernel 3: ax[row][r] = x[row] . lora_A[e_sel(chunk)][:, r]  (rank-16, fp32)
// ---------------------------------------------------------------------------
__global__ __launch_bounds__(256) void lora_ax_kernel(
    const float* __restrict__ x, const float* __restrict__ lora_A,
    const int* __restrict__ expert_idx, float* __restrict__ ax)
{
    const int chunk = blockIdx.x;
    const int e = expert_idx[chunk];
    const float* A = lora_A + (size_t)e * D_IN * RANK;
    const int tid = threadIdx.x;
    const int r  = tid & 15;
    const int rb = tid >> 4;            // 0..15
    #pragma unroll
    for (int j = 0; j < 8; ++j) {
        const int rowl = rb + 16 * j;   // 0..127
        const size_t row = (size_t)chunk * CHUNK + rowl;
        const float* xr = x + row * D_IN;
        float acc = 0.f;
        for (int k = 0; k < D_IN; ++k) acc += xr[k] * A[(size_t)k * RANK + r];
        ax[row * RANK + r] = acc;
    }
}

// ---------------------------------------------------------------------------
// Kernel 4: out = x @ W_base + b_base + SCALING * (ax @ B_e), fused epilogue.
// bf16x3 split emulation of fp32 GEMM (hi*hi + hi*lo + lo*hi), LDS-staged,
// double-buffered. Grid: (NCHUNK, D_OUT/128). Block: 256 = 8 waves (2M x 4N).
// ---------------------------------------------------------------------------
__global__ __launch_bounds__(256) void base_lora_gemm_kernel(
    const float* __restrict__ x, const float* __restrict__ Wb,
    const float* __restrict__ bb, const float* __restrict__ lora_B,
    const float* __restrict__ ax, const int* __restrict__ expert_idx,
    float* __restrict__ out)
{
    const int gm  = blockIdx.x;          // chunk (M tile == CHUNK)
    const int gn  = blockIdx.y;
    const int tid = threadIdx.x;
    const int lane  = tid & 31;
    const int wave  = tid >> 5;
    const int wm    = wave >> 2;
    const int wn    = wave & 3;
    const int lhalf = lane >> 4;
    const int lmod  = lane & 15;

    __shared__ u32 xs_hi[2][128][XSW];   // 10 KB each
    __shared__ u32 xs_lo[2][128][XSW];
    __shared__ u32 wt_hi[2][128][XSW];   // W tile transposed: [col][k]
    __shared__ u32 wt_lo[2][128][XSW];
    __shared__ float axs[CHUNK][RANK];
    __shared__ float lbs[RANK][128];

    const int e = expert_idx[gm];
    for (int i = tid; i < CHUNK * RANK; i += 256) {
        axs[i >> 4][i & 15] = ax[(size_t)gm * CHUNK * RANK + i];
        lbs[i >> 7][i & 127] =
            lora_B[(size_t)e * RANK * D_OUT + (size_t)(i >> 7) * D_OUT + gn * 128 + (i & 127)];
    }

    auto stage = [&](int nb, int kk) {
        {   // x tile: thread -> (row, 16-wide k half), split hi/lo
            const int row = tid >> 1;
            const int kh  = (tid & 1) * 16;
            const float* p = x + (size_t)(gm * CHUNK + row) * D_IN + kk + kh;
            f4 f0 = *(const f4*)(p);
            f4 f1 = *(const f4*)(p + 4);
            f4 f2v = *(const f4*)(p + 8);
            f4 f3 = *(const f4*)(p + 12);
            u32x4 h0, h1, l0, l1;
            h0[0] = pack_hi2(f0[0], f0[1]); l0[0] = pack_lo2(f0[0], f0[1]);
            h0[1] = pack_hi2(f0[2], f0[3]); l0[1] = pack_lo2(f0[2], f0[3]);
            h0[2] = pack_hi2(f1[0], f1[1]); l0[2] = pack_lo2(f1[0], f1[1]);
            h0[3] = pack_hi2(f1[2], f1[3]); l0[3] = pack_lo2(f1[2], f1[3]);
            h1[0] = pack_hi2(f2v[0], f2v[1]); l1[0] = pack_lo2(f2v[0], f2v[1]);
            h1[1] = pack_hi2(f2v[2], f2v[3]); l1[1] = pack_lo2(f2v[2], f2v[3]);
            h1[2] = pack_hi2(f3[0], f3[1]); l1[2] = pack_lo2(f3[0], f3[1]);
            h1[3] = pack_hi2(f3[2], f3[3]); l1[3] = pack_lo2(f3[2], f3[3]);
            *(u32x4*)&xs_hi[nb][row][kh >> 1]       = h0;
            *(u32x4*)&xs_hi[nb][row][(kh >> 1) + 4] = h1;
            *(u32x4*)&xs_lo[nb][row][kh >> 1]       = l0;
            *(u32x4*)&xs_lo[nb][row][(kh >> 1) + 4] = l1;
        }
        {   // W tile: thread -> 4x4 micro-block, stored transposed [col][k]
            const int kb = tid & 7;
            const int cb = tid >> 3;
            const float* wp = Wb + (size_t)(kk + kb * 4) * D_OUT + gn * 128 + cb * 4;
            f4 w0 = *(const f4*)(wp);
            f4 w1 = *(const f4*)(wp + D_OUT);
            f4 w2 = *(const f4*)(wp + 2 * D_OUT);
            f4 w3 = *(const f4*)(wp + 3 * D_OUT);
            #pragma unroll
            for (int j = 0; j < 4; ++j) {
                u32x2 h, l;
                h[0] = pack_hi2(w0[j], w1[j]); l[0] = pack_lo2(w0[j], w1[j]);
                h[1] = pack_hi2(w2[j], w3[j]); l[1] = pack_lo2(w2[j], w3[j]);
                *(u32x2*)&wt_hi[nb][cb * 4 + j][kb * 2] = h;
                *(u32x2*)&wt_lo[nb][cb * 4 + j][kb * 2] = l;
            }
        }
    };

    v8f acc[4][2];
    #pragma unroll
    for (int a = 0; a < 4; ++a)
        #pragma unroll
        for (int b = 0; b < 2; ++b)
            #pragma unroll
            for (int i = 0; i < 8; ++i) acc[a][b][i] = 0.f;

    stage(0, 0);

    for (int k0 = 0; k0 < D_IN; k0 += 32) {
        const int cbuf = (k0 >> 5) & 1;
        __syncthreads();
        if (k0 + 32 < D_IN) stage(cbuf ^ 1, k0 + 32);

        frag_u BH[2], BL[2];
        #pragma unroll
        for (int tn = 0; tn < 2; ++tn) {
            const int col = wn * 32 + tn * 16 + lmod;
            BH[tn].q[0] = *(const u32x4*)&wt_hi[cbuf][col][lhalf * 8];
            BH[tn].q[1] = *(const u32x4*)&wt_hi[cbuf][col][lhalf * 8 + 4];
            BL[tn].q[0] = *(const u32x4*)&wt_lo[cbuf][col][lhalf * 8];
            BL[tn].q[1] = *(const u32x4*)&wt_lo[cbuf][col][lhalf * 8 + 4];
        }
        #pragma unroll
        for (int tm = 0; tm < 4; ++tm) {
            const int row = wm * 64 + tm * 16 + lmod;
            frag_u AH, AL;
            AH.q[0] = *(const u32x4*)&xs_hi[cbuf][row][lhalf * 8];
            AH.q[1] = *(const u32x4*)&xs_hi[cbuf][row][lhalf * 8 + 4];
            AL.q[0] = *(const u32x4*)&xs_lo[cbuf][row][lhalf * 8];
            AL.q[1] = *(const u32x4*)&xs_lo[cbuf][row][lhalf * 8 + 4];
            #pragma unroll
            for (int tn = 0; tn < 2; ++tn) {
                acc[tm][tn] = __builtin_amdgcn_wmma_f32_16x16x32_bf16(
                    false, AH.v, false, BH[tn].v, (short)0, acc[tm][tn], false, false);
                acc[tm][tn] = __builtin_amdgcn_wmma_f32_16x16x32_bf16(
                    false, AH.v, false, BL[tn].v, (short)0, acc[tm][tn], false, false);
                acc[tm][tn] = __builtin_amdgcn_wmma_f32_16x16x32_bf16(
                    false, AL.v, false, BH[tn].v, (short)0, acc[tm][tn], false, false);
            }
        }
    }

    // Epilogue: bias + rank-16 LoRA outer product from LDS.
    #pragma unroll
    for (int tm = 0; tm < 4; ++tm) {
        #pragma unroll
        for (int i = 0; i < 8; ++i) {
            const int rowl = wm * 64 + tm * 16 + lhalf * 8 + i;
            float ar[RANK];
            #pragma unroll
            for (int r = 0; r < RANK; ++r) ar[r] = axs[rowl][r];
            #pragma unroll
            for (int tn = 0; tn < 2; ++tn) {
                const int lcol = wn * 32 + tn * 16 + lmod;
                const int col  = gn * 128 + lcol;
                float lora = 0.f;
                #pragma unroll
                for (int r = 0; r < RANK; ++r) lora += ar[r] * lbs[r][lcol];
                const size_t row = (size_t)gm * CHUNK + rowl;
                out[row * D_OUT + col] = acc[tm][tn][i] + bb[col] + SCALING * lora;
            }
        }
    }
}

// ---------------------------------------------------------------------------
extern "C" void kernel_launch(void* const* d_in, const int* in_sizes, int n_in,
                              void* d_out, int out_size, void* d_ws, size_t ws_size,
                              hipStream_t stream) {
    const float* x      = (const float*)d_in[0];
    const float* W_base = (const float*)d_in[1];
    const float* b_base = (const float*)d_in[2];
    const float* W1     = (const float*)d_in[3];
    const float* b1     = (const float*)d_in[4];
    const float* W2     = (const float*)d_in[5];
    const float* b2     = (const float*)d_in[6];
    const float* lora_A = (const float*)d_in[7];
    const float* lora_B = (const float*)d_in[8];
    float* out = (float*)d_out;

    float* chunkSum   = (float*)d_ws;                       // 64 * 2048 f32
    float* ax         = chunkSum + (size_t)NCHUNK * D_HID;  // 8192 * 16 f32
    int*   expert_idx = (int*)(ax + (size_t)NTOK * RANK);   // 64 i32

    router_hidden_kernel<<<dim3(NCHUNK, D_HID / 128), 256, 0, stream>>>(x, W1, b1, chunkSum);
    router_logits_scan_kernel<<<1, 256, 0, stream>>>(chunkSum, W2, b2, expert_idx);
    lora_ax_kernel<<<NCHUNK, 256, 0, stream>>>(x, lora_A, expert_idx, ax);
    base_lora_gemm_kernel<<<dim3(NCHUNK, D_OUT / 128), 256, 0, stream>>>(
        x, W_base, b_base, lora_B, ax, expert_idx, out);
}
